// GNN_37615323578488
// MI455X (gfx1250) — compile-verified, matrix operand order checked by reading, not run on hardware
//
#include <hip/hip_runtime.h>
#include <hip/hip_bf16.h>
#include <math.h>

// Problem constants (match reference)
#define NNODES 50000
#define NEDGES 800000
#define INFEAT 128
#define NH 8
#define EMB 32
#define OUTF 16
#define EFD 7
#define QKVW 640   // 256 (q) + 256 (k) + 128 (v)

typedef __attribute__((ext_vector_type(16))) __bf16 v16bf;
typedef __attribute__((ext_vector_type(8)))  __bf16 v8bf;
typedef __attribute__((ext_vector_type(8)))  float  v8f;

// Monotonic float <-> uint encoding so atomicMax(u32) implements float max
__device__ __forceinline__ unsigned encf(float f) {
  unsigned u = __builtin_bit_cast(unsigned, f);
  return (u & 0x80000000u) ? ~u : (u | 0x80000000u);
}
__device__ __forceinline__ float decf(unsigned u) {
  unsigned b = (u & 0x80000000u) ? (u ^ 0x80000000u) : ~u;
  return __builtin_bit_cast(float, b);
}
#define ENC_NEG_INF 0x007FFFFFu   // encf(-inf)

// ---------- kernel 0: init workspace ----------
__global__ void gnn_init(unsigned* __restrict__ mEnc,
                         float* __restrict__ denom,
                         float* __restrict__ accum,
                         int nN) {
  int i = blockIdx.x * blockDim.x + threadIdx.x;
  if (i < nN * NH * OUTF) accum[i] = 0.0f;
  if (i < nN * NH) { mEnc[i] = ENC_NEG_INF; denom[i] = 0.0f; }
}

// ---------- kernel 0b: convert feat f32 -> bf16 (native cvt) ----------
__global__ void gnn_conv_feat(const float* __restrict__ feat,
                              __bf16* __restrict__ featbf, int n) {
  int i = blockIdx.x * blockDim.x + threadIdx.x;
  if (i < n) featbf[i] = (__bf16)feat[i];
}

// ---------- kernel 0c: fuse + transpose weights -> bf16 [640 cols][128 K] ----------
__global__ void gnn_conv_w(const float* __restrict__ Wq,
                           const float* __restrict__ Wk,
                           const float* __restrict__ Wv,
                           __bf16* __restrict__ wtbf) {
  int i = blockIdx.x * blockDim.x + threadIdx.x;
  if (i >= QKVW * INFEAT) return;
  const int c = i >> 7;       // fused output column 0..639
  const int k = i & 127;      // K index
  float v;
  if (c < 256)      v = Wq[(size_t)k * 256 + c];
  else if (c < 512) v = Wk[(size_t)k * 256 + (c - 256)];
  else              v = Wv[(size_t)k * 128 + (c - 512)];
  wtbf[i] = (__bf16)v;
}

// ---------- kernel 1: QKV projection via WMMA (bf16 in, f32 acc) ----------
// One wave computes a 16x16 tile of the [N, 640] fused output.
// Inner loop: 2x b128 A loads + 2x b128 B loads + 1 wmma per k-step.
__global__ __launch_bounds__(256)
void gnn_qkv_gemm(const __bf16* __restrict__ featbf,
                  const __bf16* __restrict__ wtbf,
                  float* __restrict__ qkv,
                  int nN) {
  const int lane  = threadIdx.x & 31;
  const int wave  = threadIdx.x >> 5;         // 8 waves per block
  const int mbase = blockIdx.x * 16;
  const int ntile = blockIdx.y * 8 + wave;    // 0..39
  const int ncol  = ntile * 16;

  const int r    = lane & 15;
  const int hi   = lane >> 4;       // half-wave selector
  const int koff = hi * 8;          // CDNA5 16-bit A/B K striping

  int arowi = mbase + r; if (arowi >= nN) arowi = nN - 1;   // clamp (N%16==0 normally)
  const __bf16* arow = featbf + (size_t)arowi * INFEAT;
  const __bf16* brow = wtbf + (size_t)(ncol + r) * INFEAT;  // transposed weights

  union { v8f v; float e[8]; } c;
  #pragma unroll
  for (int i = 0; i < 8; ++i) c.e[i] = 0.0f;

  #pragma unroll
  for (int kb = 0; kb < INFEAT; kb += 32) {
    // lanes 0-15: K = {kb+0..7, kb+16..23}; lanes 16-31: K = {kb+8..15, kb+24..31}
    union { v16bf v; v8bf h[2]; } a, b;
    a.h[0] = *(const v8bf*)(arow + kb + koff);
    a.h[1] = *(const v8bf*)(arow + kb + koff + 16);
    b.h[0] = *(const v8bf*)(brow + kb + koff);
    b.h[1] = *(const v8bf*)(brow + kb + koff + 16);
    c.v = __builtin_amdgcn_wmma_f32_16x16x32_bf16(
        /*neg_a=*/false, a.v, /*neg_b=*/false, b.v,
        /*c_mod=*/(short)0, c.v, /*reuse_a=*/false, /*reuse_b=*/false);
  }

  // C layout: lane -> col (lane&15); VGPR g -> row g + 8*(lane>=16)
  float* obase = qkv + (size_t)(mbase + hi * 8) * QKVW + ncol + r;
  if (mbase + 16 <= nN) {
    // full tile fast path: straight-line stores, no per-row exec masking
    #pragma unroll
    for (int g = 0; g < 8; ++g) obase[(size_t)g * QKVW] = c.e[g];
  } else {
    #pragma unroll
    for (int g = 0; g < 8; ++g) {
      if (mbase + hi * 8 + g < nN) obase[(size_t)g * QKVW] = c.e[g];
    }
  }
}

// ---------- kernel 2: edge logits + segment max ----------
// One thread per (edge, head). ek computed on the fly from bond x Wek (LDS).
__global__ __launch_bounds__(256)
void gnn_edge_logits(const float* __restrict__ bond,
                     const int* __restrict__ src,
                     const int* __restrict__ dst,
                     const float* __restrict__ qkv,
                     const float* __restrict__ Wek,
                     const float* __restrict__ bek,
                     float* __restrict__ ebuf,
                     unsigned* __restrict__ mEnc,
                     int nE) {
  __shared__ float sW[EFD * NH * EMB];   // 7 x 256
  __shared__ float sB[NH * EMB];         // 256
  for (int i = threadIdx.x; i < EFD * NH * EMB; i += blockDim.x) sW[i] = Wek[i];
  for (int i = threadIdx.x; i < NH * EMB; i += blockDim.x)       sB[i] = bek[i];
  __syncthreads();

  int idx = blockIdx.x * blockDim.x + threadIdx.x;
  if (idx >= nE * NH) return;
  const int e = idx >> 3;
  const int h = idx & 7;
  const int s = src[e];
  const int d = dst[e];

  float bf[EFD];
  #pragma unroll
  for (int f = 0; f < EFD; ++f) bf[f] = bond[(size_t)e * EFD + f];

  const float* qrow = qkv + (size_t)s * QKVW + h * EMB;          // q block
  const float* krow = qkv + (size_t)d * QKVW + 256 + h * EMB;    // k block

  float acc = 0.0f;
  #pragma unroll
  for (int dd = 0; dd < EMB; ++dd) {
    const int col = h * EMB + dd;
    float ek = sB[col];
    #pragma unroll
    for (int f = 0; f < EFD; ++f) ek = fmaf(bf[f], sW[f * (NH * EMB) + col], ek);
    acc = fmaf(qrow[dd] + krow[dd], ek, acc);
  }
  ebuf[idx] = acc;
  atomicMax(&mEnc[(size_t)d * NH + h], encf(acc));
}

// ---------- kernel 3: exp, denom, unnormalized message scatter ----------
__global__ __launch_bounds__(256)
void gnn_edge_msg(const float* __restrict__ bond,
                  const int* __restrict__ src,
                  const int* __restrict__ dst,
                  const float* __restrict__ qkv,
                  const float* __restrict__ Wefc,
                  const float* __restrict__ befc,
                  const float* __restrict__ ebuf,
                  const unsigned* __restrict__ mEnc,
                  float* __restrict__ denom,
                  float* __restrict__ accum,
                  int nE) {
  __shared__ float sW[EFD * NH * OUTF];  // 7 x 128
  __shared__ float sB[NH * OUTF];        // 128
  for (int i = threadIdx.x; i < EFD * NH * OUTF; i += blockDim.x) sW[i] = Wefc[i];
  for (int i = threadIdx.x; i < NH * OUTF; i += blockDim.x)       sB[i] = befc[i];
  __syncthreads();

  int idx = blockIdx.x * blockDim.x + threadIdx.x;
  if (idx >= nE * NH) return;
  const int e = idx >> 3;
  const int h = idx & 7;
  const int s = src[e];
  const int d = dst[e];

  const float m  = decf(mEnc[(size_t)d * NH + h]);
  const float ex = __expf(ebuf[idx] - m);
  atomicAdd(&denom[(size_t)d * NH + h], ex);

  float bf[EFD];
  #pragma unroll
  for (int f = 0; f < EFD; ++f) bf[f] = bond[(size_t)e * EFD + f];

  const float* vrow = qkv + (size_t)s * QKVW + 512 + h * OUTF;   // v block
  float* arow = accum + (size_t)d * (NH * OUTF) + h * OUTF;
  #pragma unroll
  for (int o = 0; o < OUTF; ++o) {
    const int col = h * OUTF + o;
    float ef = sB[col];
    #pragma unroll
    for (int f = 0; f < EFD; ++f) ef = fmaf(bf[f], sW[f * (NH * OUTF) + col], ef);
    atomicAdd(&arow[o], vrow[o] * ef * ex);
  }
}

// ---------- kernel 4: normalize + bias ----------
__global__ void gnn_finalize(const float* __restrict__ accum,
                             const float* __restrict__ denom,
                             const float* __restrict__ bias,
                             float* __restrict__ out,
                             int nN) {
  int i = blockIdx.x * blockDim.x + threadIdx.x;
  if (i >= nN * NH * OUTF) return;
  const int n = i >> 7;          // /128
  const int c = i & 127;
  const int h = c >> 4;          // /16
  const float dn = denom[(size_t)n * NH + h];
  const float v  = (dn > 0.0f) ? (accum[i] / dn) : 0.0f;
  out[i] = v + bias[c];
}

// ---------- launcher ----------
extern "C" void kernel_launch(void* const* d_in, const int* in_sizes, int n_in,
                              void* d_out, int out_size, void* d_ws, size_t ws_size,
                              hipStream_t stream) {
  const float* feat = (const float*)d_in[0];
  const float* bond = (const float*)d_in[1];
  const int*   src  = (const int*)d_in[2];
  const int*   dst  = (const int*)d_in[3];
  const float* Wq   = (const float*)d_in[4];
  const float* Wk   = (const float*)d_in[5];
  const float* Wv   = (const float*)d_in[6];
  const float* Wek  = (const float*)d_in[7];
  const float* bek  = (const float*)d_in[8];
  const float* Wefc = (const float*)d_in[9];
  const float* befc = (const float*)d_in[10];
  const float* bias = (const float*)d_in[11];
  float* out = (float*)d_out;

  const int nN = in_sizes[0] / INFEAT;   // 50000
  const int nE = in_sizes[2];            // 800000

  // workspace carve-up (keep 256B alignment for b128 vector loads)
  char* ws = (char*)d_ws;
  size_t off = 0;
  auto align256 = [](size_t v) { return (v + 255) & ~(size_t)255; };
  float*    qkv    = (float*)(ws + off);  off = align256(off + (size_t)nN * QKVW * sizeof(float));
  __bf16*   featbf = (__bf16*)(ws + off); off = align256(off + (size_t)nN * INFEAT * sizeof(__bf16));
  __bf16*   wtbf   = (__bf16*)(ws + off); off = align256(off + (size_t)QKVW * INFEAT * sizeof(__bf16));
  unsigned* mEnc   = (unsigned*)(ws + off); off = align256(off + (size_t)nN * NH * sizeof(unsigned));
  float*    denom  = (float*)(ws + off);  off = align256(off + (size_t)nN * NH * sizeof(float));
  float*    accum  = (float*)(ws + off);  off = align256(off + (size_t)nN * NH * OUTF * sizeof(float));
  float*    ebuf   = (float*)(ws + off);  off = align256(off + (size_t)nE * NH * sizeof(float));
  (void)ws_size; (void)n_in; (void)out_size;

  const int TB = 256;
  const int nNodeElems = nN * NH * OUTF;     // 6.4M
  const int nEdgeWork  = nE * NH;            // 6.4M
  const int nFeatElems = nN * INFEAT;        // 6.4M
  const int nWElems    = QKVW * INFEAT;      // 81920

  gnn_init<<<(nNodeElems + TB - 1) / TB, TB, 0, stream>>>(mEnc, denom, accum, nN);
  gnn_conv_feat<<<(nFeatElems + TB - 1) / TB, TB, 0, stream>>>(feat, featbf, nFeatElems);
  gnn_conv_w<<<(nWElems + TB - 1) / TB, TB, 0, stream>>>(Wq, Wk, Wv, wtbf);

  dim3 ggrid((nN + 15) / 16, 5);             // 3125 M-tiles x (40 N-tiles / 8 waves)
  gnn_qkv_gemm<<<ggrid, TB, 0, stream>>>(featbf, wtbf, qkv, nN);

  gnn_edge_logits<<<(nEdgeWork + TB - 1) / TB, TB, 0, stream>>>(
      bond, src, dst, qkv, Wek, bek, ebuf, mEnc, nE);

  gnn_edge_msg<<<(nEdgeWork + TB - 1) / TB, TB, 0, stream>>>(
      bond, src, dst, qkv, Wefc, befc, ebuf, mEnc, denom, accum, nE);

  gnn_finalize<<<(nNodeElems + TB - 1) / TB, TB, 0, stream>>>(accum, denom, bias, out, nN);
}
